// GQA10M_16690242912259
// MI455X (gfx1250) — compile-verified, matrix operand order checked by reading, not run on hardware
//
#include <hip/hip_runtime.h>
#include <hip/hip_bf16.h>


typedef __attribute__((ext_vector_type(8)))  int      v8i;
typedef __attribute__((ext_vector_type(8)))  float    v8f;
typedef __attribute__((ext_vector_type(16))) _Float16 v16h;

#define HIDDEN 640
#define LSEQ   512
#define BATCH  32
#define MTOK   (BATCH * LSEQ)
#define NQKV   896
#define HD     64
#define NQH    10
#define LN5E5  13.122363377404328f

__device__ __forceinline__ unsigned lds_addr_of(const void* p) {
  return (unsigned)(unsigned long long)(
      __attribute__((address_space(3))) const void*)p;
}

__device__ __forceinline__ void async_copy16(const void* gsrc, const void* ldst) {
  const unsigned loff = lds_addr_of(ldst);
  const unsigned long long gaddr = (unsigned long long)(size_t)gsrc;
  asm volatile("global_load_async_to_lds_b128 %0, %1, off"
               :: "v"(loff), "v"(gaddr)
               : "memory");
}

__global__ void quant_weights_kernel(const float* __restrict__ qw,
                                     const float* __restrict__ kw,
                                     const float* __restrict__ vw,
                                     const float* __restrict__ ow,
                                     signed char* __restrict__ wqkv,
                                     signed char* __restrict__ wo,
                                     float* __restrict__ alphas) {
  const int mat = blockIdx.x;
  const float* src;
  signed char* dst;
  int rows;
  if (mat == 0)      { src = qw; dst = wqkv;               rows = 640; }
  else if (mat == 1) { src = kw; dst = wqkv + 640 * 640;   rows = 128; }
  else if (mat == 2) { src = vw; dst = wqkv + 768 * 640;   rows = 128; }
  else               { src = ow; dst = wo;                 rows = 640; }
  const int n = rows * HIDDEN;

  __shared__ float red[256];
  float s = 0.f;
  for (int i = threadIdx.x; i < n; i += 256) s += fabsf(src[i]);
  red[threadIdx.x] = s;
  __syncthreads();
  for (int t = 128; t > 0; t >>= 1) {
    if (threadIdx.x < t) red[threadIdx.x] += red[threadIdx.x + t];
    __syncthreads();
  }
  const float alpha = fmaxf(red[0] / (float)n, 1e-10f);
  if (threadIdx.x == 0) alphas[mat] = alpha;
  const float inv = 1.0f / alpha;
  for (int i = threadIdx.x; i < n; i += 256) {
    float w = rintf(src[i] * inv);
    w = fminf(1.f, fmaxf(-1.f, w));
    dst[i] = (signed char)(int)w;
  }
}

__global__ void act_quant_kernel(const float* __restrict__ x,
                                 const float* __restrict__ w1,
                                 const float* __restrict__ g,
                                 signed char* __restrict__ outq,
                                 float* __restrict__ outscale,
                                 int double_norm) {
  const int tok = blockIdx.x;
  const int tid = threadIdx.x;
  const float* xr = x + (size_t)tok * HIDDEN;
  __shared__ float red[128];

  float v[5];
  float ss = 0.f;
#pragma unroll
  for (int j = 0; j < 5; ++j) { v[j] = xr[tid * 5 + j]; ss += v[j] * v[j]; }
  red[tid] = ss;
  __syncthreads();
  for (int t = 64; t > 0; t >>= 1) {
    if (tid < t) red[tid] += red[tid + t];
    __syncthreads();
  }
  float inv = 1.0f / sqrtf(red[0] * (1.0f / HIDDEN) + 1e-6f);
  __syncthreads();

  if (double_norm) {
#pragma unroll
    for (int j = 0; j < 5; ++j) v[j] = v[j] * inv * w1[tid * 5 + j];
    float ss2 = 0.f;
#pragma unroll
    for (int j = 0; j < 5; ++j) ss2 += v[j] * v[j];
    red[tid] = ss2;
    __syncthreads();
    for (int t = 64; t > 0; t >>= 1) {
      if (tid < t) red[tid] += red[tid + t];
      __syncthreads();
    }
    inv = 1.0f / sqrtf(red[0] * (1.0f / HIDDEN) + 1e-6f);
    __syncthreads();
  }
#pragma unroll
  for (int j = 0; j < 5; ++j) v[j] = v[j] * inv * g[tid * 5 + j];

  float amax = 0.f;
#pragma unroll
  for (int j = 0; j < 5; ++j) amax = fmaxf(amax, fabsf(v[j]));
  red[tid] = amax;
  __syncthreads();
  for (int t = 64; t > 0; t >>= 1) {
    if (tid < t) red[tid] = fmaxf(red[tid], red[tid + t]);
    __syncthreads();
  }
  const float gamma = fmaxf(red[0], 1e-10f);
  const float qs = 127.0f / gamma;
#pragma unroll
  for (int j = 0; j < 5; ++j) {
    float q = fminf(127.f, fmaxf(-128.f, rintf(v[j] * qs)));
    outq[(size_t)tok * HIDDEN + tid * 5 + j] = (signed char)(int)q;
  }
  if (tid == 0) outscale[tok] = gamma * (1.0f / 127.0f);
}


__global__ void gemm_qkv_kernel(const signed char* __restrict__ actq,
                                const signed char* __restrict__ wq,
                                const float* __restrict__ alphas,
                                const float* __restrict__ tokscale,
                                _Float16* __restrict__ qkv) {
  const int lane = threadIdx.x;
  const int ln = lane & 15, hh = lane >> 4;
  const int tm = blockIdx.x, tn = blockIdx.y;
  const int n = tn * 16 + ln;

  const signed char* ap = actq + (size_t)(tm * 16 + ln) * HIDDEN;
  const signed char* bp = wq   + (size_t)n * HIDDEN;

  v8i acc = {};
#pragma unroll
  for (int kb = 0; kb < 10; ++kb) {
    if (kb < 9) {
      __builtin_prefetch(ap + (kb + 1) * 64, 0, 3);
      __builtin_prefetch(bp + (kb + 1) * 64, 0, 3);
    }
    v8i A, Bv;
#pragma unroll
    for (int vv = 0; vv < 8; vv += 2) {
      const int kbase = kb * 64 + (vv >> 1) * 16 + hh * 8;
      int2 da = *(const int2*)(ap + kbase);
      int2 db = *(const int2*)(bp + kbase);
      A[vv] = da.x;  A[vv + 1] = da.y;
      Bv[vv] = db.x; Bv[vv + 1] = db.y;
    }
    acc = __builtin_amdgcn_wmma_i32_16x16x64_iu8(true, A, true, Bv, acc,
                                                 false, false);
  }

  const int n0 = tn * 16;
  const float alpha = alphas[(n0 < 640) ? 0 : ((n0 < 768) ? 1 : 2)];
  const bool do_rope = (n0 < 768);
  float fr = 0.f;
  if (do_rope) fr = __expf(-LN5E5 * (float)(n & 62) * (1.0f / HD));

  float vals[8];
#pragma unroll
  for (int r = 0; r < 8; ++r) {
    const int mrow = tm * 16 + r + hh * 8;
    vals[r] = (float)acc[r] * alpha * tokscale[mrow];
  }
  if (do_rope) {
#pragma unroll
    for (int r = 0; r < 8; ++r) {
      const float other = __shfl_xor(vals[r], 1, 32);
      const int mrow = tm * 16 + r + hh * 8;
      const float ang = (float)(mrow & (LSEQ - 1)) * fr;
      float sn, cs;
      __sincosf(ang, &sn, &cs);
      vals[r] = (n & 1) ? fmaf(other, sn, vals[r] * cs)
                        : fmaf(-other, sn, vals[r] * cs);
    }
  }
#pragma unroll
  for (int r = 0; r < 8; ++r) {
    const int mrow = tm * 16 + r + hh * 8;
    qkv[(size_t)mrow * NQKV + n] = (_Float16)vals[r];
  }
}

__global__ void attention_kernel(const _Float16* __restrict__ qkv,
                                 float* __restrict__ attnout) {
  const int tid = threadIdx.x;
  const int lane = tid & 31;
  const int wv = tid >> 5;
  const int ln = lane & 15, hh = lane >> 4;
  const int bh = blockIdx.y;
  const int b = bh / NQH, qh = bh - b * NQH;
  const int kvh = qh / 5;
  const int q0 = blockIdx.x * 64 + wv * 16;

  const _Float16* base = qkv + (size_t)b * LSEQ * NQKV;

  __shared__ _Float16 kbuf[2][32][64];
  __shared__ _Float16 vbuf[2][32][64];
  __shared__ _Float16 pbuf[4][16][32];

  const char* kg = (const char*)(base + 640 + kvh * HD);
  const char* vg = (const char*)(base + 768 + kvh * HD);

  auto issue_chunk = [&](int chunk) {
    const int key0 = chunk * 32;
    _Float16(*kb)[64] = kbuf[chunk & 1];
    _Float16(*vb)[64] = vbuf[chunk & 1];
#pragma unroll
    for (int j = 0; j < 4; ++j) {
      const int i = j * 128 + tid;
      const int sel = i >> 8;
      const int r = (i >> 3) & 31;
      const int c = i & 7;
      const char* src =
          (sel ? vg : kg) + (size_t)(key0 + r) * (NQKV * 2) + c * 16;
      const _Float16* dst = (sel ? vb : kb)[r] + c * 8;
      async_copy16(src, dst);
    }
  };

  v16h qa0 = {}, qa1 = {};
  {
    const _Float16* qp = base + (size_t)(q0 + ln) * NQKV + qh * HD;
#pragma unroll
    for (int v = 0; v < 8; ++v) {
      const int kk = (v >> 2) * 16 + hh * 8 + (v & 3) * 2;
      qa0[2 * v] = qp[kk];      qa0[2 * v + 1] = qp[kk + 1];
      qa1[2 * v] = qp[32 + kk]; qa1[2 * v + 1] = qp[32 + kk + 1];
    }
  }

  v8f acc[4] = {v8f{}, v8f{}, v8f{}, v8f{}};
  float rowm[8], rowl[8];
#pragma unroll
  for (int r = 0; r < 8; ++r) { rowm[r] = -1e30f; rowl[r] = 0.f; }

  issue_chunk(0);

  for (int kt2 = 0; kt2 < 16; ++kt2) {
    const int cur = kt2 & 1;
    if (kt2 + 1 < 16) {
      issue_chunk(kt2 + 1);
      asm volatile("s_wait_asynccnt 4" ::: "memory");
    } else {
      asm volatile("s_wait_asynccnt 0" ::: "memory");
    }
    __syncthreads();

    v8f s0 = {}, s1 = {};
    {
      const _Float16* kp = kbuf[cur][ln];
      v16h kb0 = {}, kb1 = {};
#pragma unroll
      for (int v = 0; v < 8; ++v) {
        const int kk = (v >> 2) * 16 + hh * 8 + (v & 3) * 2;
        kb0[2 * v] = kp[kk];      kb0[2 * v + 1] = kp[kk + 1];
        kb1[2 * v] = kp[32 + kk]; kb1[2 * v + 1] = kp[32 + kk + 1];
      }
      s0 = __builtin_amdgcn_wmma_f32_16x16x32_f16(false, qa0, false, kb0,
                                                  (short)0, s0, false, false);
      s0 = __builtin_amdgcn_wmma_f32_16x16x32_f16(false, qa1, false, kb1,
                                                  (short)0, s0, false, false);
    }
    {
      const _Float16* kp = kbuf[cur][16 + ln];
      v16h kb0 = {}, kb1 = {};
#pragma unroll
      for (int v = 0; v < 8; ++v) {
        const int kk = (v >> 2) * 16 + hh * 8 + (v & 3) * 2;
        kb0[2 * v] = kp[kk];      kb0[2 * v + 1] = kp[kk + 1];
        kb1[2 * v] = kp[32 + kk]; kb1[2 * v + 1] = kp[32 + kk + 1];
      }
      s1 = __builtin_amdgcn_wmma_f32_16x16x32_f16(false, qa0, false, kb0,
                                                  (short)0, s1, false, false);
      s1 = __builtin_amdgcn_wmma_f32_16x16x32_f16(false, qa1, false, kb1,
                                                  (short)0, s1, false, false);
    }

#pragma unroll
    for (int r = 0; r < 8; ++r) {
      const float a0 = s0[r] * 0.125f;
      const float a1 = s1[r] * 0.125f;
      float tmax = fmaxf(a0, a1);
      for (int m = 1; m < 16; m <<= 1) tmax = fmaxf(tmax, __shfl_xor(tmax, m, 16));
      const float mnew = fmaxf(rowm[r], tmax);
      const float corr = __expf(rowm[r] - mnew);
      const float p0 = __expf(a0 - mnew);
      const float p1 = __expf(a1 - mnew);
      float psum = p0 + p1;
      for (int m = 1; m < 16; m <<= 1) psum += __shfl_xor(psum, m, 16);
      rowl[r] = rowl[r] * corr + psum;
      rowm[r] = mnew;
      acc[0][r] *= corr; acc[1][r] *= corr; acc[2][r] *= corr; acc[3][r] *= corr;
      pbuf[wv][r + hh * 8][ln]      = (_Float16)p0;
      pbuf[wv][r + hh * 8][ln + 16] = (_Float16)p1;
    }

    v16h pa = {};
#pragma unroll
    for (int v = 0; v < 8; ++v) {
      const int kk = (v >> 2) * 16 + hh * 8 + (v & 3) * 2;
      pa[2 * v]     = pbuf[wv][ln][kk];
      pa[2 * v + 1] = pbuf[wv][ln][kk + 1];
    }

#pragma unroll
    for (int g = 0; g < 4; ++g) {
      v16h vb = {};
#pragma unroll
      for (int v = 0; v < 8; ++v) {
        const int kk = (v >> 2) * 16 + hh * 8 + (v & 3) * 2;
        vb[2 * v]     = vbuf[cur][kk][g * 16 + ln];
        vb[2 * v + 1] = vbuf[cur][kk + 1][g * 16 + ln];
      }
      acc[g] = __builtin_amdgcn_wmma_f32_16x16x32_f16(false, pa, false, vb,
                                                      (short)0, acc[g],
                                                      false, false);
    }
    __syncthreads();
  }

#pragma unroll
  for (int r = 0; r < 8; ++r) {
    const int qrow = q0 + r + hh * 8;
    const float invl = 1.0f / rowl[r];
    float* o = attnout + ((size_t)(b * LSEQ + qrow)) * HIDDEN + qh * HD;
    o[ln]      = acc[0][r] * invl;
    o[16 + ln] = acc[1][r] * invl;
    o[32 + ln] = acc[2][r] * invl;
    o[48 + ln] = acc[3][r] * invl;
  }
}

__global__ void gemm_out_kernel(const signed char* __restrict__ actq,
                                const signed char* __restrict__ wq,
                                const float* __restrict__ alphas,
                                const float* __restrict__ tokscale,
                                const float* __restrict__ residual,
                                float* __restrict__ out) {
  const int lane = threadIdx.x;
  const int ln = lane & 15, hh = lane >> 4;
  const int tm = blockIdx.x, tn = blockIdx.y;
  const int n = tn * 16 + ln;

  const signed char* ap = actq + (size_t)(tm * 16 + ln) * HIDDEN;
  const signed char* bp = wq   + (size_t)n * HIDDEN;

  v8i acc = {};
#pragma unroll
  for (int kb = 0; kb < 10; ++kb) {
    if (kb < 9) {
      __builtin_prefetch(ap + (kb + 1) * 64, 0, 3);
      __builtin_prefetch(bp + (kb + 1) * 64, 0, 3);
    }
    v8i A, Bv;
#pragma unroll
    for (int vv = 0; vv < 8; vv += 2) {
      const int kbase = kb * 64 + (vv >> 1) * 16 + hh * 8;
      int2 da = *(const int2*)(ap + kbase);
      int2 db = *(const int2*)(bp + kbase);
      A[vv] = da.x;  A[vv + 1] = da.y;
      Bv[vv] = db.x; Bv[vv + 1] = db.y;
    }
    acc = __builtin_amdgcn_wmma_i32_16x16x64_iu8(true, A, true, Bv, acc,
                                                 false, false);
  }

  const float alpha = alphas[3];
#pragma unroll
  for (int r = 0; r < 8; ++r) {
    const int mrow = tm * 16 + r + hh * 8;
    const size_t idx = (size_t)mrow * HIDDEN + n;
    out[idx] = (float)acc[r] * alpha * tokscale[mrow] + residual[idx];
  }
}

extern "C" void kernel_launch(void* const* d_in, const int* in_sizes, int n_in,
                              void* d_out, int out_size, void* d_ws,
                              size_t ws_size, hipStream_t stream) {
  (void)in_sizes; (void)n_in; (void)out_size; (void)ws_size;
  const float* x      = (const float*)d_in[0];
  const float* norm_w = (const float*)d_in[1];
  const float* q_w    = (const float*)d_in[2];
  const float* q_g    = (const float*)d_in[3];
  const float* k_w    = (const float*)d_in[4];
  const float* v_w    = (const float*)d_in[6];
  const float* o_w    = (const float*)d_in[8];
  const float* o_g    = (const float*)d_in[9];
  float* out = (float*)d_out;

  char* p = (char*)d_ws;
  auto take = [&](size_t bytes) -> char* {
    char* r = p;
    p += (bytes + 255) & ~(size_t)255;
    return r;
  };
  signed char* wqkv    = (signed char*)take((size_t)NQKV * HIDDEN);
  signed char* wo      = (signed char*)take((size_t)HIDDEN * HIDDEN);
  float*       alphas  = (float*)take(4 * sizeof(float));
  signed char* actq    = (signed char*)take((size_t)MTOK * HIDDEN);
  float*       tokscale= (float*)take((size_t)MTOK * sizeof(float));
  _Float16*    qkvh    = (_Float16*)take((size_t)MTOK * NQKV * 2);
  float*       attnout = (float*)take((size_t)MTOK * HIDDEN * 4);
  signed char* atq     = (signed char*)take((size_t)MTOK * HIDDEN);
  float*       oscale  = (float*)take((size_t)MTOK * sizeof(float));

  quant_weights_kernel<<<4, 256, 0, stream>>>(q_w, k_w, v_w, o_w, wqkv, wo,
                                              alphas);
  act_quant_kernel<<<MTOK, 128, 0, stream>>>(x, norm_w, q_g, actq, tokscale, 1);
  gemm_qkv_kernel<<<dim3(MTOK / 16, NQKV / 16), 32, 0, stream>>>(
      actq, wqkv, alphas, tokscale, qkvh);
  attention_kernel<<<dim3(LSEQ / 64, BATCH * NQH), 128, 0, stream>>>(qkvh,
                                                                     attnout);
  act_quant_kernel<<<MTOK, 128, 0, stream>>>(attnout, nullptr, o_g, atq,
                                             oscale, 0);
  gemm_out_kernel<<<dim3(MTOK / 16, HIDDEN / 16), 32, 0, stream>>>(
      atq, wo, alphas, oscale, x, out);
}
